// LinearHijack_62663572848780
// MI455X (gfx1250) — matrix operand order repaired and checked by reading, not compiled, not run on hardware
//
#include <hip/hip_runtime.h>
#include <hip/hip_bf16.h>

#define MDIM 8192
#define NDIM 8192
#define BATCH 4
#define KSPLIT 8
#define SEGN (NDIM / KSPLIT)          // 1024 K-elements per segment
#define WAVES_PER_WG 8
#define TILE_M 16
#define MPW (WAVES_PER_WG * TILE_M)   // 128 m-rows per workgroup

typedef __attribute__((ext_vector_type(16))) _Float16 v16h;
typedef __attribute__((ext_vector_type(8)))  float    v8f;
typedef __attribute__((ext_vector_type(4)))  int      v4i;
typedef __attribute__((ext_vector_type(2)))  _Float16 h2;

union H2U { h2 h; unsigned int u; };
union V16 { v16h v; unsigned int u[8]; v4i q[2]; };

// ---------------------------------------------------------------------------
// Main kernel: dequant fp4 -> f16 and WMMA-accumulate one K-segment.
// grid = (MDIM/MPW, KSPLIT), block = 256 (8 waves of 32).
// Each wave: 16x16 f32 accumulator, D = x_tile(16x32 f16) * W_tile^T(32x16 f16)
// iterated SEGN/32 = 32 times.
// ---------------------------------------------------------------------------
__global__ __launch_bounds__(256) void fp4_linear_wmma(
    const float* __restrict__ x,        // [4, NDIM] f32
    const int*   __restrict__ qw,       // [MDIM*NDIM/2] one packed byte per int32
    const float* __restrict__ absmax,   // [MDIM*NDIM/64]
    const float* __restrict__ code,     // [16] fp4 codebook
    float*       __restrict__ part)     // [KSPLIT][BATCH][MDIM] partial sums
{
  __shared__ __align__(16) _Float16 xlds[5 * SEGN];  // rows 0..3 = x, row 4 = zeros
  __shared__ unsigned int code2[256];                // byte -> packed half2 codebook

  const int tid   = threadIdx.x;
  const int seg   = blockIdx.y;
  const int nseg0 = seg * SEGN;

  // Build byte->half2 codebook: element 2j = hi nibble (even K -> low half),
  // element 2j+1 = lo nibble (odd K -> high half).
  for (int v = tid; v < 256; v += 256) {
    H2U cu;
    cu.h.x = (_Float16)code[(v >> 4) & 15];
    cu.h.y = (_Float16)code[v & 15];
    code2[v] = cu.u;
  }
  // Zero row (read by A-lanes whose padded batch row >= 4).
  for (int i = tid; i < SEGN; i += 256) xlds[4 * SEGN + i] = (_Float16)0.0f;
  // Stage this segment of x as f16.
  for (int i = tid; i < BATCH * SEGN; i += 256) {
    const int b = i >> 10;            // i / SEGN
    const int n = i & (SEGN - 1);
    xlds[b * SEGN + n] = (_Float16)x[b * NDIM + nseg0 + n];
  }
  __syncthreads();

  const int lane  = tid & 31;
  const int wave  = tid >> 5;
  const int ncol  = lane & 15;        // B column (= output row m offset) and A row M
  const int khalf = lane >> 4;        // lane half selects K sub-range
  const int m     = blockIdx.x * MPW + wave * TILE_M + ncol;
  const int arow  = (ncol < BATCH) ? ncol : BATCH;     // rows >=4 read the zero row
  const _Float16* __restrict__ xrow = xlds + arow * SEGN + khalf * 16;

  // Lane's B data: 16 consecutive W elements of row m starting at K-offset khalf*16.
  const int  ibase0 = m * NDIM + nseg0 + khalf * 16;   // flat element index (fits i32)
  const int* __restrict__ qp = qw + (ibase0 >> 1);     // 8 consecutive packed bytes

  v8f acc = {};

#pragma unroll 2
  for (int it = 0; it < SEGN / 32; ++it) {
    const int nbl = it * 32;

    // ---- B fragment: stream 8 packed bytes (NT: single-use data) ----
    V16 bq;
    bq.q[0] = __builtin_nontemporal_load((const v4i*)(qp + (nbl >> 1)));
    bq.q[1] = __builtin_nontemporal_load((const v4i*)(qp + (nbl >> 1) + 4));

    // One absmax covers all 16 of this lane's elements (aligned 64-blocks).
    const float am = absmax[(unsigned)(ibase0 + nbl) >> 6];
    const _Float16 amh = (_Float16)am;
    h2 am2; am2.x = amh; am2.y = amh;

    V16 bfrag;
#pragma unroll
    for (int t = 0; t < 8; ++t) {
      H2U cu; cu.u = code2[bq.u[t] & 0xFF];   // ds_load_b32 LUT
      H2U r;  r.h  = cu.h * am2;              // v_pk_mul_f16
      bfrag.u[t] = r.u;
    }

    // ---- A fragment from LDS (contiguous K, same order as B: K = nbl+khalf*16+{0..15}) ----
    V16 afrag;
    afrag.q[0] = *(const v4i*)(xrow + nbl);
    afrag.q[1] = *(const v4i*)(xrow + nbl + 8);

    acc = __builtin_amdgcn_wmma_f32_16x16x32_f16(
        /*neg_a=*/false, afrag.v, /*neg_b=*/false, bfrag.v,
        /*c_mod=*/(short)0, acc, /*reuse_a=*/false, /*reuse_b=*/false);
  }

  // D layout: lanes 0-15 hold N=ncol, VGPR v -> batch row M=v. Rows 0..3 are real.
  if (lane < 16) {
    float* p = part + (seg * BATCH) * MDIM + m;
    p[0]           = acc[0];
    p[MDIM]        = acc[1];
    p[2 * MDIM]    = acc[2];
    p[3 * MDIM]    = acc[3];
  }
}

// ---------------------------------------------------------------------------
// Deterministic K-segment reduction + bias add.
// ---------------------------------------------------------------------------
__global__ __launch_bounds__(256) void reduce_bias(
    const float* __restrict__ part, const float* __restrict__ bias,
    float* __restrict__ out)
{
  const int t = blockIdx.x * blockDim.x + threadIdx.x;
  if (t >= BATCH * MDIM) return;
  const int m = t & (MDIM - 1);
  const int b = t >> 13;              // t / MDIM
  float s = bias[m];
#pragma unroll
  for (int sgi = 0; sgi < KSPLIT; ++sgi)
    s += part[(sgi * BATCH + b) * MDIM + m];
  out[t] = s;
}

extern "C" void kernel_launch(void* const* d_in, const int* in_sizes, int n_in,
                              void* d_out, int out_size, void* d_ws, size_t ws_size,
                              hipStream_t stream) {
  const float* x      = (const float*)d_in[0];
  const int*   qw     = (const int*)d_in[1];
  const float* absmax = (const float*)d_in[2];
  const float* code   = (const float*)d_in[3];
  const float* bias   = (const float*)d_in[4];
  float* out  = (float*)d_out;
  float* part = (float*)d_ws;         // KSPLIT*BATCH*MDIM f32 = 1 MB

  dim3 grid(MDIM / MPW, KSPLIT);      // (64, 8) workgroups, 4096 waves
  fp4_linear_wmma<<<grid, 256, 0, stream>>>(x, qw, absmax, code, part);
  reduce_bias<<<(BATCH * MDIM + 255) / 256, 256, 0, stream>>>(part, bias, out);
}
